// CrossAttention_35467839930686
// MI455X (gfx1250) — compile-verified, matrix operand order checked by reading, not run on hardware
//
#include <hip/hip_runtime.h>
#include <hip/hip_bf16.h>

typedef __attribute__((ext_vector_type(16))) _Float16 v16h;
typedef __attribute__((ext_vector_type(8)))  float    v8f;
typedef __attribute__((ext_vector_type(2)))  _Float16 h2;

#define HD 1024   // hidden dim
#define NH 16     // heads
#define DH 64     // head dim
#define LQ 2048   // sequence length
#define NB 2      // batch

__device__ __forceinline__ v8f wmma16x16x32(v16h a, v16h b, v8f c) {
  // D = A(16x32 f16) * B(32x16 f16) + C(16x16 f32)
  return __builtin_amdgcn_wmma_f32_16x16x32_f16(
      /*neg_a=*/false, a, /*neg_b=*/false, b,
      /*c_mod=*/(short)0, c, /*reuse_a=*/false, /*reuse_b=*/false);
}

__device__ __forceinline__ v8f zero8() {
  v8f z;
#pragma unroll
  for (int i = 0; i < 8; ++i) z[i] = 0.0f;
  return z;
}

// Load a 16x32 f16 fragment from contiguous 32-halfword rows.
// A-layout: lane = M row, VGPR i packs K pairs {0,2,4,6,16,18,20,22}+8*half.
// Same pattern loads B fragments from N-major (pre-transposed) storage.
__device__ __forceinline__ v16h load_frag_row(const _Float16* row, int halfl) {
  v16h f;
#pragma unroll
  for (int i = 0; i < 8; ++i) {
    const int kk = ((i < 4) ? (2 * i) : (16 + 2 * (i - 4))) + 8 * halfl;
    h2 p = *(const h2*)(row + kk);
    f[2 * i]     = p.x;
    f[2 * i + 1] = p.y;
  }
  return f;
}

// ---- CDNA5 async global->LDS staging (ASYNCcnt-tracked, §08) ---------------
__device__ __forceinline__ void async_b128(unsigned lds_off, const void* gptr) {
  // GV mode: vdst = per-lane LDS byte offset, vaddr = 64-bit global address.
  asm volatile("global_load_async_to_lds_b128 %0, %1, off"
               :: "v"(lds_off), "v"((unsigned long long)gptr)
               : "memory");
}
__device__ __forceinline__ void wait_async_le4() {
  asm volatile("s_wait_asynccnt 4" ::: "memory");  // prev tile done, next in flight
}
__device__ __forceinline__ void wait_async_0() {
  asm volatile("s_wait_asynccnt 0" ::: "memory");
}

// ---------------------------------------------------------------------------
// Prep: f32 -> f16 elementwise (x, y). 4 elements/thread, coalesced.
// ---------------------------------------------------------------------------
__global__ __launch_bounds__(256)
void cvt_f16(const float* __restrict__ in, _Float16* __restrict__ out) {
  const size_t i = ((size_t)blockIdx.x * 256 + threadIdx.x) * 4;
  const float4 v = *(const float4*)(in + i);
  h2 lo, hi;
  lo.x = (_Float16)v.x; lo.y = (_Float16)v.y;
  hi.x = (_Float16)v.z; hi.y = (_Float16)v.w;
  *(h2*)(out + i)     = lo;
  *(h2*)(out + i + 2) = hi;
}

// ---------------------------------------------------------------------------
// Prep: weight convert + transpose: Wt[n*K + k] = (f16)W[k*ldw + n].
// 32x32 tiles via LDS, both sides coalesced. Grid (N/32, K/32), block (32,8).
// ---------------------------------------------------------------------------
__global__ __launch_bounds__(256)
void transpose_cvt(const float* __restrict__ W, int ldw,
                   _Float16* __restrict__ Wt, int Ksz) {
  __shared__ _Float16 tile[32][33];
  const int n0 = blockIdx.x * 32, k0 = blockIdx.y * 32;
  const int tx = threadIdx.x, ty = threadIdx.y;
#pragma unroll
  for (int j = 0; j < 4; ++j) {
    const int k = ty + j * 8;
    tile[k][tx] = (_Float16)W[(size_t)(k0 + k) * ldw + (n0 + tx)];
  }
  __syncthreads();
#pragma unroll
  for (int j = 0; j < 4; ++j) {
    const int n = ty + j * 8;
    Wt[(size_t)(n0 + n) * Ksz + (k0 + tx)] = tile[tx][n];
  }
}

// ---------------------------------------------------------------------------
// GEMM + bias (+scale), all-f16 operands, async double-buffered LDS staging.
// A: (M x Ksz) f16 row-major. Bt: (N x Ksz) f16 row-major (pre-transposed W).
// C[m][n] = (sum_k A[m][k]*Bt[n][k] + bias[n]) * scale.
// Block 128 (4 waves); tile 64x64; K stepped by 32; 4 async b128 per thread
// per K-step (2 for A, 2 for Bt).
// ---------------------------------------------------------------------------
template <typename TOUT>
__global__ __launch_bounds__(128)
void gemm_f16_wmma(const _Float16* __restrict__ A,
                   const _Float16* __restrict__ Bt,
                   const float* __restrict__ bias,
                   TOUT* __restrict__ C, int ldc, int Ksz, float scale) {
  __shared__ __align__(16) _Float16 As[2][64 * 32];  // M-major, pitch 32
  __shared__ __align__(16) _Float16 Bs[2][64 * 32];  // N-major, pitch 32

  const int tid   = threadIdx.x;
  const int wave  = tid >> 5;
  const int lane  = tid & 31;
  const int halfl = lane >> 4;
  const int nl    = lane & 15;

  const int row0 = blockIdx.y * 64;
  const int col0 = blockIdx.x * 64;

  const unsigned as0 = (unsigned)(unsigned long long)&As[0][0];
  const unsigned bs0 = (unsigned)(unsigned long long)&Bs[0][0];

  v8f acc[4];
#pragma unroll
  for (int t = 0; t < 4; ++t) acc[t] = zero8();

  auto issue = [&](int buf, int k0) {
    for (int c = tid; c < 256; c += 128) {       // 16B chunks: 4 per 32-f16 row
      const int r = c >> 2, cc = c & 3;
      async_b128(as0 + buf * 4096 + r * 64 + cc * 16,
                 A + (size_t)(row0 + r) * Ksz + k0 + cc * 8);
      async_b128(bs0 + buf * 4096 + r * 64 + cc * 16,
                 Bt + (size_t)(col0 + r) * Ksz + k0 + cc * 8);
    }
  };

  const int nk = Ksz / 32;
  issue(0, 0);
  for (int t = 0; t < nk; ++t) {
    if (t + 1 < nk) {
      issue((t + 1) & 1, (t + 1) * 32);
      wait_async_le4();                          // own tile-t copies landed
    } else {
      wait_async_0();
    }
    __syncthreads();                             // everyone's tile-t data in LDS

    const _Float16* Asb = &As[t & 1][0];
    const _Float16* Bsb = &Bs[t & 1][0];
    const v16h a = load_frag_row(&Asb[(wave * 16 + nl) * 32], halfl);
#pragma unroll
    for (int t4 = 0; t4 < 4; ++t4) {
      const v16h b = load_frag_row(&Bsb[(t4 * 16 + nl) * 32], halfl);
      acc[t4] = wmma16x16x32(a, b, acc[t4]);
    }
    __syncthreads();                             // done reading before buf reuse
  }

#pragma unroll
  for (int t = 0; t < 4; ++t) {
#pragma unroll
    for (int r = 0; r < 8; ++r) {
      const int row = row0 + wave * 16 + r + 8 * halfl;  // C layout: VGPR r -> M=r+8*half
      const int col = col0 + t * 16 + nl;                //           lane  -> N
      const float v = (acc[t][r] + bias[col]) * scale;
      C[(size_t)row * ldc + col] = (TOUT)v;
    }
  }
}

// ---------------------------------------------------------------------------
// Flash attention, causal + key-padding mask. Q has 1/sqrt(D) pre-folded.
// Grid (L/64, B*H); block 128 (4 waves x 16 q-rows). Per 32-key tile:
// async-stage K to LDS, register-transpose V to LDS (overlaps async),
// S = Q*K^T (4 WMMAs), online softmax, O += P*V (4 WMMAs).
// ---------------------------------------------------------------------------
__global__ __launch_bounds__(128)
void flash_attn_wmma(const _Float16* __restrict__ Q,
                     const _Float16* __restrict__ K,
                     const _Float16* __restrict__ V,
                     const unsigned char* __restrict__ mask,
                     _Float16* __restrict__ O) {
  __shared__ __align__(16) _Float16 Ks[32 * 64];      // key-major rows of 64 features
  __shared__ __align__(16) _Float16 Vt[64 * 34];      // d-major (transposed), pitch 34
  __shared__ __align__(16) _Float16 Pb[4 * 16 * 32];  // per-wave P relayout scratch

  const int tid   = threadIdx.x;
  const int wave  = tid >> 5;
  const int lane  = tid & 31;
  const int halfl = lane >> 4;
  const int nl    = lane & 15;

  const int bh = blockIdx.y;
  const int b  = bh / NH;
  const int h  = bh % NH;
  const int q0 = blockIdx.x * 64 + wave * 16;

  const size_t rowbase = (size_t)b * LQ;
  const unsigned ks0 = (unsigned)(unsigned long long)&Ks[0];

  // Q fragments for both 32-wide d-halves (scale already folded in).
  const _Float16* qrow = Q + (rowbase + q0 + nl) * HD + h * DH;
  const v16h aq0 = load_frag_row(qrow, halfl);
  const v16h aq1 = load_frag_row(qrow + 32, halfl);

  float m_i[8], l_i[8];
#pragma unroll
  for (int r = 0; r < 8; ++r) { m_i[r] = -1e30f; l_i[r] = 0.0f; }
  v8f acc[4];
#pragma unroll
  for (int t = 0; t < 4; ++t) acc[t] = zero8();

  const int ntiles = 2 * blockIdx.x + 2;  // causal: keys <= blockIdx.x*64 + 63
  const unsigned char* mrow = mask + rowbase;

  for (int kt = 0; kt < ntiles; ++kt) {
    const int key0 = kt * 32;
    __syncthreads();  // previous tile's readers done
    // Async-stage K tile: 32 keys x 64 f16 = 256 16B chunks, 2 per thread.
    for (int c = tid; c < 256; c += 128) {
      const int r = c >> 3, cc = c & 7;
      async_b128(ks0 + r * 128 + cc * 16,
                 K + (rowbase + key0 + r) * HD + h * DH + cc * 8);
    }
    // Register-transpose V tile into Vt[d][key] while K copies fly.
    for (int c = tid; c < 32 * 32; c += 128) {
      const int r = c >> 5, d2 = (c & 31) * 2;
      const h2 p = *(const h2*)(V + (rowbase + key0 + r) * HD + h * DH + d2);
      Vt[d2 * 34 + r]       = p.x;
      Vt[(d2 + 1) * 34 + r] = p.y;
    }
    wait_async_0();
    __syncthreads();

    // Scores: K rows are B-fragments directly (B[k][n] = K[n][k]).
    v8f s0 = zero8(), s1 = zero8();
    {
      const _Float16* kr0 = &Ks[nl * 64];
      const _Float16* kr1 = &Ks[(16 + nl) * 64];
      s0 = wmma16x16x32(aq0, load_frag_row(kr0, halfl), s0);
      s0 = wmma16x16x32(aq1, load_frag_row(kr0 + 32, halfl), s0);
      s1 = wmma16x16x32(aq0, load_frag_row(kr1, halfl), s1);
      s1 = wmma16x16x32(aq1, load_frag_row(kr1 + 32, halfl), s1);
    }

    const int keyA = key0 + nl;
    const int keyB = keyA + 16;
    const bool padA = mrow[keyA] != 0;
    const bool padB = mrow[keyB] != 0;

    _Float16* Pw = &Pb[wave * 16 * 32];

#pragma unroll
    for (int r = 0; r < 8; ++r) {
      const int qidx = q0 + r + 8 * halfl;
      const float v0 = (keyA > qidx || padA) ? -1e30f : s0[r];
      const float v1 = (keyB > qidx || padB) ? -1e30f : s1[r];
      // Row reductions across the 16 lanes of this half (xor<16 stays in-half).
      float rm = fmaxf(v0, v1);
#pragma unroll
      for (int xm = 1; xm < 16; xm <<= 1)
        rm = fmaxf(rm, __shfl_xor(rm, xm, 32));
      const float nm   = fmaxf(m_i[r], rm);
      const float corr = __expf(m_i[r] - nm);
      m_i[r] = nm;
      const float p0 = __expf(v0 - nm);
      const float p1 = __expf(v1 - nm);
      float rs = p0 + p1;
#pragma unroll
      for (int xm = 1; xm < 16; xm <<= 1)
        rs += __shfl_xor(rs, xm, 32);
      l_i[r] = l_i[r] * corr + rs;
#pragma unroll
      for (int t = 0; t < 4; ++t) acc[t][r] *= corr;
      // Spill P in C-layout; reread in A-layout (same-wave LDS is ordered).
      const int prow = r + 8 * halfl;
      Pw[prow * 32 + nl]      = (_Float16)p0;
      Pw[prow * 32 + 16 + nl] = (_Float16)p1;
    }

    // O += P(16x32) * V(32x64): Vt columns are B-fragments.
    const v16h ap = load_frag_row(&Pw[nl * 32], halfl);
#pragma unroll
    for (int t = 0; t < 4; ++t) {
      const v16h bv = load_frag_row(&Vt[(t * 16 + nl) * 34], halfl);
      acc[t] = wmma16x16x32(ap, bv, acc[t]);
    }
  }

  // Normalize and write attention output in (B*L, HD) row-major f16.
#pragma unroll
  for (int t = 0; t < 4; ++t) {
#pragma unroll
    for (int r = 0; r < 8; ++r) {
      const int row = q0 + r + 8 * halfl;
      const int col = h * DH + t * 16 + nl;
      O[(rowbase + row) * HD + col] = (_Float16)(acc[t][r] / l_i[r]);
    }
  }
}

extern "C" void kernel_launch(void* const* d_in, const int* in_sizes, int n_in,
                              void* d_out, int out_size, void* d_ws, size_t ws_size,
                              hipStream_t stream) {
  const float*         x      = (const float*)d_in[0];
  const float*         y      = (const float*)d_in[1];
  const unsigned char* mask   = (const unsigned char*)d_in[2];
  const float*         Wq_w   = (const float*)d_in[3];
  const float*         Wq_b   = (const float*)d_in[4];
  const float*         Wkv_w  = (const float*)d_in[5];
  const float*         Wkv_b  = (const float*)d_in[6];
  const float*         proj_w = (const float*)d_in[7];
  const float*         proj_b = (const float*)d_in[8];
  float*               out    = (float*)d_out;

  const size_t M = (size_t)NB * LQ;  // 4096 rows
  const size_t E = M * HD;           // 4M elements

  _Float16* Xh  = (_Float16*)d_ws;   // activations f16
  _Float16* Yh  = Xh + E;
  _Float16* WqT = Yh + E;            // pre-transposed f16 weights (N x K)
  _Float16* WkT = WqT + (size_t)HD * HD;
  _Float16* WvT = WkT + (size_t)HD * HD;
  _Float16* WpT = WvT + (size_t)HD * HD;
  _Float16* Qb  = WpT + (size_t)HD * HD;
  _Float16* Kb  = Qb + E;
  _Float16* Vb  = Kb + E;
  _Float16* Ob  = Vb + E;            // total 28M f16 = 56 MB

  const dim3 blk128(128);
  const dim3 grid_g(HD / 64, M / 64);        // 16 x 64
  const dim3 grid_t(HD / 32, HD / 32);       // 32 x 32
  const dim3 blk_t(32, 8);

  // one-time prep: activations to f16; weights to f16 + transpose (N-major)
  cvt_f16<<<dim3(E / 1024), dim3(256), 0, stream>>>(x, Xh);
  cvt_f16<<<dim3(E / 1024), dim3(256), 0, stream>>>(y, Yh);
  transpose_cvt<<<grid_t, blk_t, 0, stream>>>(Wq_w,        HD,     WqT, HD);
  transpose_cvt<<<grid_t, blk_t, 0, stream>>>(Wkv_w,       2 * HD, WkT, HD);
  transpose_cvt<<<grid_t, blk_t, 0, stream>>>(Wkv_w + HD,  2 * HD, WvT, HD);
  transpose_cvt<<<grid_t, blk_t, 0, stream>>>(proj_w,      HD,     WpT, HD);

  // Q = (x @ Wq + b) * 1/sqrt(D)   (scale folded into Q)
  gemm_f16_wmma<_Float16><<<grid_g, blk128, 0, stream>>>(
      Xh, WqT, Wq_b, Qb, HD, HD, 0.125f);
  gemm_f16_wmma<_Float16><<<grid_g, blk128, 0, stream>>>(
      Yh, WkT, Wkv_b, Kb, HD, HD, 1.0f);
  gemm_f16_wmma<_Float16><<<grid_g, blk128, 0, stream>>>(
      Yh, WvT, Wkv_b + HD, Vb, HD, HD, 1.0f);

  flash_attn_wmma<<<dim3(LQ / 64, NB * NH), blk128, 0, stream>>>(
      Qb, Kb, Vb, mask, Ob);

  // out = attn @ proj_w + proj_b  (f32 output)
  gemm_f16_wmma<float><<<grid_g, blk128, 0, stream>>>(
      Ob, WpT, proj_b, out, HD, HD, 1.0f);
}
